// LERS_68925635166732
// MI455X (gfx1250) — matrix operand order repaired and checked by reading, not compiled, not run on hardware
//
#include <hip/hip_runtime.h>
#include <hip/hip_bf16.h>

#define DD    128
#define TT    8
#define NH    8
#define HDIM  16
#define FFD   512
#define NLAY  6
#define N_ADM 2048
#define N_LAB 753
#define N_DRG 4294
#define NE1   16384
#define NE2   16384

typedef __attribute__((ext_vector_type(16))) __bf16 v16bf;
typedef __attribute__((ext_vector_type(8)))  float  v8f;

static __device__ __forceinline__ unsigned short f2bf_bits(float f) {
  union { float f; unsigned u; } x; x.f = f;
  unsigned r = x.u + 0x7FFFu + ((x.u >> 16) & 1u);
  return (unsigned short)(r >> 16);
}

// load 8 consecutive floats (16B-aligned) as two b128s
static __device__ __forceinline__ void ld8(const float* __restrict__ p, float* d) {
  float4 a = ((const float4*)p)[0];
  float4 b = ((const float4*)p)[1];
  d[0] = a.x; d[1] = a.y; d[2] = a.z; d[3] = a.w;
  d[4] = b.x; d[5] = b.y; d[6] = b.z; d[7] = b.w;
}

// ---------------------------------------------------------------------------
// WMMA GEMM: C[M,N] = act( A[M,KC](+A2) @ B + bias + addin )
// BTRANS=0: B is [KC,N] row-major.  BTRANS=1: B is [N,KC] row-major (A@B^T).
// One wave per 16x16 C tile, K fully unrolled in steps of 32, bf16 convert.
// ---------------------------------------------------------------------------
template<int KC, bool BTRANS, bool HASA2>
__global__ __launch_bounds__(256) void k_gemm_wmma(
    const float* __restrict__ A, const float* __restrict__ A2,
    const float* __restrict__ B, const float* __restrict__ bias,
    const float* __restrict__ addin, float* __restrict__ C,
    int M, int N, int relu)
{
  __shared__ __align__(16) float Bs[8 * 512];     // 32x16 f32 tile per wave
  const int lane      = threadIdx.x & 31;
  const int waveInBlk = threadIdx.x >> 5;
  const int wave      = blockIdx.x * 8 + waveInBlk;
  const int tilesN = (N + 15) >> 4;
  const int tilesM = (M + 15) >> 4;
  if (wave >= tilesM * tilesN) return;
  const int tm = (wave / tilesN) << 4;
  const int tn = (wave % tilesN) << 4;
  float* bs = &Bs[waveInBlk * 512];

  v8f acc;
#pragma unroll
  for (int i = 0; i < 8; ++i) acc[i] = 0.f;

  const int  am   = lane & 15;
  const int  half = lane >> 4;
  const bool aok  = (tm + am) < M;
  const int  arow = aok ? (tm + am) : (M - 1);     // clamp: always-valid address
  const float* Ap  = A + (size_t)arow * KC;
  const float* A2p = HASA2 ? (A2 + (size_t)arow * KC) : nullptr;
  const bool fullN = (tn + 16) <= N;

  const int nlo = lane & 15;
  const int khi = (lane >> 4) << 4;                 // 0 or 16

#pragma unroll
  for (int k0 = 0; k0 < KC; k0 += 32) {
    // ---- A fragment (vector loads, zero-select at M edge)
    float av[16];
    ld8(Ap + k0 + half * 8,      av);
    ld8(Ap + k0 + 16 + half * 8, av + 8);
    if (HASA2) {
      float a2[16];
      ld8(A2p + k0 + half * 8,      a2);
      ld8(A2p + k0 + 16 + half * 8, a2 + 8);
#pragma unroll
      for (int j = 0; j < 16; ++j) av[j] += a2[j];
    }
    union { v16bf v; unsigned short u[16]; } af;
#pragma unroll
    for (int j = 0; j < 16; ++j) af.u[j] = f2bf_bits(aok ? av[j] : 0.f);

    // ---- stage B 32x16 tile into LDS as [k_local][n_local]
    if (!BTRANS) {
      const float* Bp = B + (size_t)(k0 + lane) * N + tn;   // row k0+lane
      if (fullN) {
        float4* dst = (float4*)(bs + lane * 16);
#pragma unroll
        for (int i = 0; i < 4; ++i) dst[i] = ((const float4*)Bp)[i];
      } else {
#pragma unroll
        for (int j = 0; j < 16; ++j) {
          bool ok = (tn + j) < N;
          bs[lane * 16 + j] = ok ? Bp[ok ? j : 0] : 0.f;
        }
      }
    } else {
      const bool nok = (tn + nlo) < N;
      const int  n   = nok ? (tn + nlo) : (N - 1);
      float bv[16];
      const float* Bp = B + (size_t)n * KC + k0 + khi;
      ld8(Bp,     bv);
      ld8(Bp + 8, bv + 8);
#pragma unroll
      for (int j = 0; j < 16; ++j)
        bs[(khi + j) * 16 + nlo] = nok ? bv[j] : 0.f;
    }
    asm volatile("s_wait_dscnt 0" ::: "memory");

    // ---- B fragment: lanes 0-15 hold K 0..7,16..23, lanes 16-31 hold K 8..15,24..31
    union { v16bf v; unsigned short u[16]; } bfr;
#pragma unroll
    for (int j = 0; j < 16; ++j)
      bfr.u[j] = f2bf_bits(bs[((j & 7) + (khi >> 1) + ((j >> 3) << 4)) * 16 + nlo]);

    acc = __builtin_amdgcn_wmma_f32_16x16x32_bf16(
        false, af.v, false, bfr.v, (short)0, acc, false, false);
  }

  // ---- store: VGPR r -> M = r (+8 for upper half), N = lane&15
  const int cn = tn + nlo;
  const int mb = (lane >> 4) << 3;
  const bool cok = cn < N;
  float bv = (bias && cok) ? bias[cn] : 0.f;
#pragma unroll
  for (int r = 0; r < 8; ++r) {
    int cm = tm + mb + r;
    if (cm < M && cok) {
      float v = acc[r] + bv;
      if (addin) v += addin[(size_t)cm * N + cn];
      if (relu)  v = v > 0.f ? v : 0.f;
      C[(size_t)cm * N + cn] = v;
    }
  }
}

// ---------------------------------------------------------------------------
// Node feature projection: h[n,d] = b[d] + sum_f x[n,f] w[f,d] (+ emb[id[n],d])
// ---------------------------------------------------------------------------
__global__ void k_proj_nodes(const float* __restrict__ x, int nf,
                             const float* __restrict__ w, const float* __restrict__ b,
                             const float* __restrict__ emb, const int* __restrict__ id,
                             float* __restrict__ h, int Nn)
{
  int gid = blockIdx.x * blockDim.x + threadIdx.x;
  if (gid >= Nn * DD) return;
  int n = gid >> 7, d = gid & 127;
  float v = b[d];
  for (int f = 0; f < nf; ++f) v += x[n * nf + f] * w[f * DD + d];
  if (emb) { int e = id ? id[n] : n; v += emb[(size_t)e * DD + d]; }
  h[gid] = v;
}

// ---------------------------------------------------------------------------
// GINE message: msg[didx[e],d] += relu(hsrc[sidx[e],d] + proj(edge_x[e])[d])
// ---------------------------------------------------------------------------
__global__ __launch_bounds__(128) void k_gine_msg(
    const float* __restrict__ hsrc, const int* __restrict__ sidx,
    const int* __restrict__ didx, const float* __restrict__ ex, int nf,
    const float* __restrict__ we, const float* __restrict__ be,
    float* __restrict__ msg)
{
  int e = blockIdx.x, d = threadIdx.x;
  float p = be[d];
  for (int f = 0; f < nf; ++f) p += ex[(size_t)e * nf + f] * we[f * DD + d];
  float v = hsrc[(size_t)sidx[e] * DD + d] + p;
  v = v > 0.f ? v : 0.f;
  atomicAdd(&msg[(size_t)didx[e] * DD + d], v);
}

// ---------------------------------------------------------------------------
// Sinusoidal PE add
// ---------------------------------------------------------------------------
__global__ void k_add_pe(const float* __restrict__ src, float* __restrict__ X,
                         int t, int Nn)
{
  int gid = blockIdx.x * blockDim.x + threadIdx.x;
  if (gid >= Nn * DD) return;
  int d = gid & 127;
  float ang = (float)t * expf(-(float)(d & ~1) * (logf(10000.f) / (float)DD));
  float pe = (d & 1) ? cosf(ang) : sinf(ang);
  X[gid] = src[gid] + pe;
}

// ---------------------------------------------------------------------------
// Causal MHA over T=8. One thread per (t_q, n, h). 64B-aligned float4 loads.
// Q from Qb cols 0..127 of 384; K/V from KVb cols 128..383.
// ---------------------------------------------------------------------------
__global__ void k_attn(const float* __restrict__ Qb, const float* __restrict__ KVb,
                       float* __restrict__ O, int Nn)
{
  int gid = blockIdx.x * blockDim.x + threadIdx.x;
  if (gid >= TT * Nn * NH) return;
  int h  = gid % NH;
  int n  = (gid / NH) % Nn;
  int tq = gid / (NH * Nn);

  float qv[HDIM];
  ld8(Qb + (size_t)(tq * Nn + n) * 384 + h * HDIM,     qv);
  ld8(Qb + (size_t)(tq * Nn + n) * 384 + h * HDIM + 8, qv + 8);

  float lg[TT], mx = -1e30f;
  for (int tk = 0; tk <= tq; ++tk) {
    float kv[HDIM];
    const float* kp = KVb + (size_t)(tk * Nn + n) * 384 + DD + h * HDIM;
    ld8(kp, kv); ld8(kp + 8, kv + 8);
    float s = 0.f;
#pragma unroll
    for (int d = 0; d < HDIM; ++d) s += qv[d] * kv[d];
    s *= 0.25f;                       // 1/sqrt(16)
    lg[tk] = s;
    mx = s > mx ? s : mx;
  }
  float den = 0.f;
  for (int tk = 0; tk <= tq; ++tk) { lg[tk] = expf(lg[tk] - mx); den += lg[tk]; }
  float inv = 1.f / den;

  float o[HDIM];
#pragma unroll
  for (int d = 0; d < HDIM; ++d) o[d] = 0.f;
  for (int tk = 0; tk <= tq; ++tk) {
    float wgt = lg[tk] * inv;
    float vv[HDIM];
    const float* vp = KVb + (size_t)(tk * Nn + n) * 384 + 2 * DD + h * HDIM;
    ld8(vp, vv); ld8(vp + 8, vv + 8);
#pragma unroll
    for (int d = 0; d < HDIM; ++d) o[d] += wgt * vv[d];
  }
  float* op = O + (size_t)(tq * Nn + n) * DD + h * HDIM;
#pragma unroll
  for (int i = 0; i < 4; ++i)
    ((float4*)op)[i] = make_float4(o[i * 4], o[i * 4 + 1], o[i * 4 + 2], o[i * 4 + 3]);
}

// ---------------------------------------------------------------------------
// Fused residual + LayerNorm (in place): X = LN(X + R) * g + b. One wave/row.
// ---------------------------------------------------------------------------
__global__ __launch_bounds__(256) void k_ln(float* __restrict__ X,
                                            const float* __restrict__ R,
                                            const float* __restrict__ g,
                                            const float* __restrict__ b, int Mrows)
{
  int lane = threadIdx.x & 31;
  int row  = blockIdx.x * 8 + (threadIdx.x >> 5);
  if (row >= Mrows) return;
  float* x = X + (size_t)row * DD;
  const float* r = R + (size_t)row * DD;
  float4 xv = ((const float4*)x)[lane];
  float4 rv = ((const float4*)r)[lane];
  float v[4] = { xv.x + rv.x, xv.y + rv.y, xv.z + rv.z, xv.w + rv.w };
  float s = v[0] + v[1] + v[2] + v[3];
#pragma unroll
  for (int off = 16; off; off >>= 1) s += __shfl_xor(s, off, 32);
  float mu = s * (1.f / 128.f);
  float s2 = 0.f;
#pragma unroll
  for (int i = 0; i < 4; ++i) { float d = v[i] - mu; s2 += d * d; }
#pragma unroll
  for (int off = 16; off; off >>= 1) s2 += __shfl_xor(s2, off, 32);
  float inv = rsqrtf(s2 * (1.f / 128.f) + 1e-5f);
  float4 gv = ((const float4*)g)[lane];
  float4 bv = ((const float4*)b)[lane];
  float4 ov = make_float4((v[0] - mu) * inv * gv.x + bv.x,
                          (v[1] - mu) * inv * gv.y + bv.y,
                          (v[2] - mu) * inv * gv.z + bv.z,
                          (v[3] - mu) * inv * gv.w + bv.w);
  ((float4*)x)[lane] = ov;
}

__global__ void k_zero(float* __restrict__ p, size_t n)
{
  size_t i = (size_t)blockIdx.x * blockDim.x + threadIdx.x;
  if (i < n) p[i] = 0.f;
}

__global__ void k_scatter1(float* __restrict__ out, const int* __restrict__ s,
                           const int* __restrict__ d, int E, int Ncols)
{
  int gid = blockIdx.x * blockDim.x + threadIdx.x;
  if (gid >= TT * E) return;
  int t = gid / E;
  out[(size_t)t * N_ADM * Ncols + (size_t)s[gid] * Ncols + d[gid]] = 1.0f;
}

// ---------------------------------------------------------------------------
static inline int cdiv(long a, long b) { return (int)((a + b - 1) / b); }

static void gemm(hipStream_t st, const float* A, const float* A2, const float* B,
                 int btrans, const float* bias, const float* addin, float* C,
                 int M, int N, int K, int relu)
{
  int waves  = cdiv(M, 16) * cdiv(N, 16);
  int blocks = cdiv(waves, 8);
  if (K == 512) {
    k_gemm_wmma<512, false, false><<<blocks, 256, 0, st>>>(A, nullptr, B, bias, addin, C, M, N, relu);
  } else if (btrans) {
    k_gemm_wmma<128, true, false><<<blocks, 256, 0, st>>>(A, nullptr, B, bias, addin, C, M, N, relu);
  } else if (A2) {
    k_gemm_wmma<128, false, true><<<blocks, 256, 0, st>>>(A, A2, B, bias, addin, C, M, N, relu);
  } else {
    k_gemm_wmma<128, false, false><<<blocks, 256, 0, st>>>(A, nullptr, B, bias, addin, C, M, N, relu);
  }
}

extern "C" void kernel_launch(void* const* d_in, const int* in_sizes, int n_in,
                              void* d_out, int out_size, void* d_ws, size_t ws_size,
                              hipStream_t stream)
{
  const float* x_adm      = (const float*)d_in[0];
  const float* x_lab      = (const float*)d_in[1];
  const float* x_drug     = (const float*)d_in[2];
  const float* ex_did     = (const float*)d_in[3];
  const float* ex_took    = (const float*)d_in[4];
  const float* w_p_adm    = (const float*)d_in[5];
  const float* b_p_adm    = (const float*)d_in[6];
  const float* w_p_lab    = (const float*)d_in[7];
  const float* b_p_lab    = (const float*)d_in[8];
  const float* w_p_drug   = (const float*)d_in[9];
  const float* b_p_drug   = (const float*)d_in[10];
  const float* w_p_e      = (const float*)d_in[11];
  const float* b_p_e      = (const float*)d_in[12];
  const float* w_p_e4d    = (const float*)d_in[13];
  const float* b_p_e4d    = (const float*)d_in[14];
  const float* emb_lab    = (const float*)d_in[15];
  const float* emb_drug   = (const float*)d_in[16];
  const float* gnn_w      = (const float*)d_in[17];
  const float* gnn_b      = (const float*)d_in[18];
  const float* sa_in_w    = (const float*)d_in[19];
  const float* sa_in_b    = (const float*)d_in[20];
  const float* sa_out_w   = (const float*)d_in[21];
  const float* sa_out_b   = (const float*)d_in[22];
  const float* ca_in_w    = (const float*)d_in[23];
  const float* ca_in_b    = (const float*)d_in[24];
  const float* ca_out_w   = (const float*)d_in[25];
  const float* ca_out_b   = (const float*)d_in[26];
  const float* ff1_w      = (const float*)d_in[27];
  const float* ff1_b      = (const float*)d_in[28];
  const float* ff2_w      = (const float*)d_in[29];
  const float* ff2_b      = (const float*)d_in[30];
  const float* ln1_w      = (const float*)d_in[31];
  const float* ln1_b      = (const float*)d_in[32];
  const float* ln2_w      = (const float*)d_in[33];
  const float* ln2_b      = (const float*)d_in[34];
  const float* ln3_w      = (const float*)d_in[35];
  const float* ln3_b      = (const float*)d_in[36];
  const int*   id_lab     = (const int*)d_in[37];
  const int*   id_drug    = (const int*)d_in[38];
  const int*   src_did    = (const int*)d_in[39];
  const int*   dst_did    = (const int*)d_in[40];
  const int*   src_took   = (const int*)d_in[41];
  const int*   dst_took   = (const int*)d_in[42];
  const int*   l_s_did    = (const int*)d_in[43];
  const int*   l_d_did    = (const int*)d_in[44];
  const int*   l_s_took   = (const int*)d_in[45];
  const int*   l_d_took   = (const int*)d_in[46];

  float* ws = (float*)d_ws;
  size_t off = 0;
  auto alloc = [&](size_t n) { float* p = ws + off; off += n; return p; };

  float* hA   = alloc((size_t)N_ADM * DD);
  float* hL   = alloc((size_t)N_LAB * DD);
  float* hDr  = alloc((size_t)N_DRG * DD);
  float* mA1  = alloc((size_t)N_ADM * DD);
  float* mA2  = alloc((size_t)N_ADM * DD);
  float* mL   = alloc((size_t)N_LAB * DD);
  float* mD   = alloc((size_t)N_DRG * DD);
  float* s1a  = alloc((size_t)N_ADM * DD);
  float* s1l  = alloc((size_t)N_LAB * DD);
  float* s1d  = alloc((size_t)N_DRG * DD);
  float* s2a  = alloc((size_t)N_ADM * DD);
  float* s2l  = alloc((size_t)N_LAB * DD);
  float* s2d  = alloc((size_t)N_DRG * DD);
  float* tmpA = alloc((size_t)N_ADM * DD);
  float* Xa   = alloc((size_t)TT * N_ADM * DD);
  float* Xl   = alloc((size_t)TT * N_LAB * DD);
  float* Xd   = alloc((size_t)TT * N_DRG * DD);
  float* X0a  = alloc((size_t)TT * N_ADM * DD);
  float* X0l  = alloc((size_t)TT * N_LAB * DD);
  float* X0d  = alloc((size_t)TT * N_DRG * DD);
  size_t Mmax = (size_t)TT * N_DRG;                 // 34352 rows
  float* qkvA = alloc(Mmax * 384);
  float* qkvB = alloc(Mmax * 384);
  float* Obuf = alloc(Mmax * DD);
  float* Tbuf = alloc(Mmax * DD);
  float* Hbuf = alloc(Mmax * FFD);

  // ---- base node projections (shared across timesteps)
  k_proj_nodes<<<cdiv((long)N_ADM * DD, 256), 256, 0, stream>>>(
      x_adm, 8, w_p_adm, b_p_adm, nullptr, nullptr, hA, N_ADM);
  k_proj_nodes<<<cdiv((long)N_LAB * DD, 256), 256, 0, stream>>>(
      x_lab, 2, w_p_lab, b_p_lab, emb_lab, id_lab, hL, N_LAB);
  k_proj_nodes<<<cdiv((long)N_DRG * DD, 256), 256, 0, stream>>>(
      x_drug, 8, w_p_drug, b_p_drug, emb_drug, id_drug, hDr, N_DRG);

  // ---- GNN: 2 GINE layers per timestep, per-t weights
  for (int t = 0; t < TT; ++t) {
    const float* exd = ex_did  + (size_t)t * NE1 * 2;
    const float* ext = ex_took + (size_t)t * NE2 * 7;
    const int *s1i = src_did + t * NE1, *d1i = dst_did + t * NE1;
    const int *s2i = src_took + t * NE2, *d2i = dst_took + t * NE2;

    const float *cA = hA, *cL = hL, *cD = hDr;
    for (int l = 0; l < 2; ++l) {
      float* oA = (l == 0) ? s1a : s2a;
      float* oL = (l == 0) ? s1l : s2l;
      float* oD = (l == 0) ? s1d : s2d;

      k_zero<<<cdiv((long)N_ADM * DD, 256), 256, 0, stream>>>(mA1, (size_t)N_ADM * DD);
      k_zero<<<cdiv((long)N_ADM * DD, 256), 256, 0, stream>>>(mA2, (size_t)N_ADM * DD);
      k_zero<<<cdiv((long)N_LAB * DD, 256), 256, 0, stream>>>(mL,  (size_t)N_LAB * DD);
      k_zero<<<cdiv((long)N_DRG * DD, 256), 256, 0, stream>>>(mD,  (size_t)N_DRG * DD);

      // adm <- lab (reverse edges), adm <- drug, lab <- adm, drug <- adm
      k_gine_msg<<<NE1, 128, 0, stream>>>(cL, d1i, s1i, exd, 2, w_p_e,   b_p_e,   mA1);
      k_gine_msg<<<NE2, 128, 0, stream>>>(cD, d2i, s2i, ext, 7, w_p_e4d, b_p_e4d, mA2);
      k_gine_msg<<<NE1, 128, 0, stream>>>(cA, s1i, d1i, exd, 2, w_p_e,   b_p_e,   mL);
      k_gine_msg<<<NE2, 128, 0, stream>>>(cA, s2i, d2i, ext, 7, w_p_e4d, b_p_e4d, mD);

      const float* W  = gnn_w + (size_t)((t * 2 + l) * 4) * DD * DD;
      const float* Bb = gnn_b + (size_t)((t * 2 + l) * 4) * DD;
      // na = relu( (mA1+ha)@W1 + b1 + (mA2+ha)@W3 + b3 )
      gemm(stream, mA1, cA, W + 1 * DD * DD, 0, Bb + 1 * DD, nullptr, tmpA, N_ADM, DD, DD, 0);
      gemm(stream, mA2, cA, W + 3 * DD * DD, 0, Bb + 3 * DD, tmpA,    oA,   N_ADM, DD, DD, 1);
      gemm(stream, mL,  cL, W + 0 * DD * DD, 0, Bb + 0 * DD, nullptr, oL,   N_LAB, DD, DD, 1);
      gemm(stream, mD,  cD, W + 2 * DD * DD, 0, Bb + 2 * DD, nullptr, oD,   N_DRG, DD, DD, 1);
      cA = oA; cL = oL; cD = oD;
    }
    k_add_pe<<<cdiv((long)N_ADM * DD, 256), 256, 0, stream>>>(cA, Xa + (size_t)t * N_ADM * DD, t, N_ADM);
    k_add_pe<<<cdiv((long)N_LAB * DD, 256), 256, 0, stream>>>(cL, Xl + (size_t)t * N_LAB * DD, t, N_LAB);
    k_add_pe<<<cdiv((long)N_DRG * DD, 256), 256, 0, stream>>>(cD, Xd + (size_t)t * N_DRG * DD, t, N_DRG);
  }

  // decoder memory = decoder input
  hipMemcpyAsync(X0a, Xa, (size_t)TT * N_ADM * DD * sizeof(float), hipMemcpyDeviceToDevice, stream);
  hipMemcpyAsync(X0l, Xl, (size_t)TT * N_LAB * DD * sizeof(float), hipMemcpyDeviceToDevice, stream);
  hipMemcpyAsync(X0d, Xd, (size_t)TT * N_DRG * DD * sizeof(float), hipMemcpyDeviceToDevice, stream);

  // ---- three transformer decoders
  float* Xs[3]  = { Xa, Xl, Xd };
  float* X0s[3] = { X0a, X0l, X0d };
  int    Ns[3]  = { N_ADM, N_LAB, N_DRG };
  for (int i = 0; i < 3; ++i) {
    float* X = Xs[i]; float* X0 = X0s[i];
    int Nn = Ns[i], Mr = TT * Nn;
    for (int l = 0; l < NLAY; ++l) {
      size_t wo = (size_t)(i * NLAY + l);
      // self-attention
      gemm(stream, X, nullptr, sa_in_w + wo * DD * 384, 0, sa_in_b + wo * 384, nullptr, qkvA, Mr, 384, DD, 0);
      k_attn<<<cdiv((long)Mr * NH, 256), 256, 0, stream>>>(qkvA, qkvA, Obuf, Nn);
      gemm(stream, Obuf, nullptr, sa_out_w + wo * DD * DD, 0, sa_out_b + wo * DD, nullptr, Tbuf, Mr, DD, DD, 0);
      k_ln<<<cdiv(Mr, 8), 256, 0, stream>>>(X, Tbuf, ln1_w + wo * DD, ln1_b + wo * DD, Mr);
      // cross-attention (memory = decoder input)
      gemm(stream, X,  nullptr, ca_in_w + wo * DD * 384, 0, ca_in_b + wo * 384, nullptr, qkvA, Mr, 384, DD, 0);
      gemm(stream, X0, nullptr, ca_in_w + wo * DD * 384, 0, ca_in_b + wo * 384, nullptr, qkvB, Mr, 384, DD, 0);
      k_attn<<<cdiv((long)Mr * NH, 256), 256, 0, stream>>>(qkvA, qkvB, Obuf, Nn);
      gemm(stream, Obuf, nullptr, ca_out_w + wo * DD * DD, 0, ca_out_b + wo * DD, nullptr, Tbuf, Mr, DD, DD, 0);
      k_ln<<<cdiv(Mr, 8), 256, 0, stream>>>(X, Tbuf, ln2_w + wo * DD, ln2_b + wo * DD, Mr);
      // feed-forward
      gemm(stream, X,    nullptr, ff1_w + wo * DD * FFD, 0, ff1_b + wo * FFD, nullptr, Hbuf, Mr, FFD, DD, 1);
      gemm(stream, Hbuf, nullptr, ff2_w + wo * FFD * DD, 0, ff2_b + wo * DD,  nullptr, Tbuf, Mr, DD, FFD, 0);
      k_ln<<<cdiv(Mr, 8), 256, 0, stream>>>(X, Tbuf, ln3_w + wo * DD, ln3_b + wo * DD, Mr);
    }
  }

  // ---- scores + labels into d_out (concatenated in return order)
  float* out = (float*)d_out;
  size_t S1 = (size_t)TT * N_ADM * N_LAB;           // labels offset
  size_t S2 = 2 * S1;                               // scores4drug offset
  size_t S3 = S2 + (size_t)TT * N_ADM * N_DRG;      // labels4drug offset
  size_t LEN_D = (size_t)TT * N_ADM * N_DRG;

  for (int t = 0; t < TT; ++t) {
    gemm(stream, Xa + (size_t)t * N_ADM * DD, nullptr, Xl + (size_t)t * N_LAB * DD, 1,
         nullptr, nullptr, out + (size_t)t * N_ADM * N_LAB, N_ADM, N_LAB, DD, 0);
    gemm(stream, Xa + (size_t)t * N_ADM * DD, nullptr, Xd + (size_t)t * N_DRG * DD, 1,
         nullptr, nullptr, out + S2 + (size_t)t * N_ADM * N_DRG, N_ADM, N_DRG, DD, 0);
  }
  k_zero<<<cdiv((long)S1, 256), 256, 0, stream>>>(out + S1, S1);
  k_zero<<<cdiv((long)LEN_D, 256), 256, 0, stream>>>(out + S3, LEN_D);
  k_scatter1<<<cdiv((long)TT * NE1, 256), 256, 0, stream>>>(out + S1, l_s_did, l_d_did, NE1, N_LAB);
  k_scatter1<<<cdiv((long)TT * NE2, 256), 256, 0, stream>>>(out + S3, l_s_took, l_d_took, NE2, N_DRG);
}
